// SpecBuNNConv_5875515261232
// MI455X (gfx1250) — matrix operand order repaired and checked by reading, not compiled, not run on hardware
//
#include <hip/hip_runtime.h>
#include <hip/hip_bf16.h>

#define Nn 65536
#define Gg 64
#define NPG 1024
#define Kk 32
#define NB 64
#define BD 4
#define CH 256
#define NBP 384
#define Ee 524288

typedef __attribute__((ext_vector_type(16))) __bf16 bf16x16;
typedef __attribute__((ext_vector_type(8)))  float  floatx8;

union FragBF { bf16x16 v; unsigned int u[8]; };

__device__ __forceinline__ float gelu_f(float v) {
    return 0.5f * v * (1.0f + erff(v * 0.70710678118654752f));
}

// ---------------------------------------------------------------------------
// Generic bf16 WMMA GEMM: C[M,N] = act(A[M,K]@W[K,N] + bias) + residual
// A row-major stride lda (bf16), W row-major stride ldw (bf16).
// Block = 256 threads (8 waves), tile 256x64; wave w owns rows [w*32, w*32+32)
// and all 64 cols: 2x4 WMMA tiles = 8 v_wmma per K-step per wave.
// ---------------------------------------------------------------------------
__launch_bounds__(256)
__global__ void wmma_gemm_bf16_kernel(
    const __bf16* __restrict__ A, int lda,
    const __bf16* __restrict__ W, int ldw,
    const float*  __restrict__ bias,
    const float*  __restrict__ residual, int ldr,
    float*        __restrict__ C, int ldc,
    __bf16*       __restrict__ Cb, int ldcb,
    int M, int Ndim, int Kdim, int act)
{
    (void)M; (void)Ndim;
    __shared__ __bf16 sA[256][40];  // stride 40 halfwords: 16B-aligned rows, conflict-free frags
    __shared__ __bf16 sB[64][40];   // transposed [n][k]

    const int tid  = threadIdx.x;
    const int wave = tid >> 5;      // 0..7 -> M sub-tile of 32 rows
    const int lane = tid & 31;
    const int fl   = lane & 15;
    const int fh   = lane >> 4;
    const int block_m = blockIdx.x * 256;
    const int block_n = blockIdx.y * 64;

    floatx8 acc[2][4] = {};

    const int bk = tid >> 3;              // B k   0..31
    const int bn = (tid & 7) * 8;         // B n   0..56

    for (int k0 = 0; k0 < Kdim; k0 += 32) {
        // ---- stage A tile: 256x32 bf16, one full row per thread (4x uint4) ----
        {
            const __bf16* src = A + (size_t)(block_m + tid) * lda + k0;
            uint4 d0 = ((const uint4*)src)[0];
            uint4 d1 = ((const uint4*)src)[1];
            uint4 d2 = ((const uint4*)src)[2];
            uint4 d3 = ((const uint4*)src)[3];
            *(uint4*)&sA[tid][0]  = d0;
            *(uint4*)&sA[tid][8]  = d1;
            *(uint4*)&sA[tid][16] = d2;
            *(uint4*)&sA[tid][24] = d3;
            if (k0 + 32 < Kdim) __builtin_prefetch(src + 32, 0, 3);
        }
        // ---- stage B tile (transposed): W[k0+bk][block_n+bn..+7] -> sB[n][k] ----
        {
            const __bf16* src = W + (size_t)(k0 + bk) * ldw + block_n + bn;
            uint4 d = *(const uint4*)src;
            const __bf16* e = (const __bf16*)&d;
#pragma unroll
            for (int i = 0; i < 8; ++i) sB[bn + i][bk] = e[i];
            if (k0 + 32 < Kdim) __builtin_prefetch(src + (size_t)32 * ldw, 0, 3);
        }
        __syncthreads();

        // ---- fragments ----
        FragBF afrag[2], bfrag[4];
#pragma unroll
        for (int mi = 0; mi < 2; ++mi) {
            int m = wave * 32 + mi * 16 + fl;
#pragma unroll
            for (int j = 0; j < 8; ++j)
                afrag[mi].u[j] = *(const unsigned int*)&sA[m][fh * 16 + 2 * j];
        }
#pragma unroll
        for (int ni = 0; ni < 4; ++ni) {
            int n = ni * 16 + fl;
#pragma unroll
            for (int j = 0; j < 8; ++j)
                bfrag[ni].u[j] = *(const unsigned int*)&sB[n][fh * 16 + 2 * j];
        }

#pragma unroll
        for (int mi = 0; mi < 2; ++mi)
#pragma unroll
            for (int ni = 0; ni < 4; ++ni)
                acc[mi][ni] = __builtin_amdgcn_wmma_f32_16x16x32_bf16(
                    false, afrag[mi].v, false, bfrag[ni].v,
                    (short)0, acc[mi][ni], false, false);
        __syncthreads();
    }

    // ---- epilogue: lane holds (m = i + 8*fh, n = fl) of each 16x16 tile ----
#pragma unroll
    for (int mi = 0; mi < 2; ++mi) {
#pragma unroll
        for (int ni = 0; ni < 4; ++ni) {
            int n = block_n + ni * 16 + fl;
            float bv = bias ? bias[n] : 0.0f;
#pragma unroll
            for (int i = 0; i < 8; ++i) {
                int m = block_m + wave * 32 + mi * 16 + fh * 8 + i;
                float v = acc[mi][ni][i] + bv;
                if (act) v = gelu_f(v);
                if (residual) v += residual[(size_t)m * ldr + n];
                C[(size_t)m * ldc + n] = v;
                if (Cb) Cb[(size_t)m * ldcb + n] = (__bf16)v;
            }
        }
    }
}

// ---------------------------------------------------------------------------
// Elementwise helpers
// ---------------------------------------------------------------------------
__launch_bounds__(256)
__global__ void zero_f32_kernel(float* __restrict__ p, long n) {
    long i = (long)blockIdx.x * 256 + threadIdx.x;
    if (i < n) p[i] = 0.0f;
}

__launch_bounds__(256)
__global__ void convert_f32_bf16_kernel(const float* __restrict__ src, long spitch,
                                        __bf16* __restrict__ dst, long dpitch,
                                        long rows, long cols) {
    long i = (long)blockIdx.x * 256 + threadIdx.x;
    if (i >= rows * cols) return;
    long r = i / cols, c = i - r * cols;
    dst[r * dpitch + c] = (__bf16)src[r * spitch + c];
}

__launch_bounds__(256)
__global__ void bias_sum_kernel(const float* __restrict__ a, const float* __restrict__ b,
                                float* __restrict__ o, int n) {
    int i = blockIdx.x * 256 + threadIdx.x;
    if (i < n) o[i] = a[i] + b[i];
}

// agg[dst] += h[src] over all edges; 64 threads per edge, float4 per thread
__launch_bounds__(256)
__global__ void scatter_add_kernel(const int* __restrict__ src, const int* __restrict__ dst,
                                   const float* __restrict__ h, float* __restrict__ agg) {
    long idx = (long)blockIdx.x * 256 + threadIdx.x;   // E*64 total
    if (idx >= (long)Ee * 64) return;
    int e = (int)(idx >> 6);
    int c = (int)(idx & 63) << 2;
    int s = src[e], d = dst[e];
    float4 v = *(const float4*)(h + (size_t)s * CH + c);
    float* o = agg + (size_t)d * CH + c;
    atomicAdd(o + 0, v.x); atomicAdd(o + 1, v.y);
    atomicAdd(o + 2, v.z); atomicAdd(o + 3, v.w);
}

// ---------------------------------------------------------------------------
// Householder Q (4x4 from 6 params) and bundle transforms
// ---------------------------------------------------------------------------
__device__ __forceinline__ void make_Q(const float p[6], float Q[4][4]) {
#pragma unroll
    for (int i = 0; i < 4; ++i)
#pragma unroll
        for (int j = 0; j < 4; ++j) Q[i][j] = (i == j) ? 1.0f : 0.0f;
    int idx = 0;
#pragma unroll
    for (int j = 0; j < 3; ++j) {
        float v[4] = {0.f, 0.f, 0.f, 0.f};
        v[j] = 1.0f;
#pragma unroll
        for (int t = 0; t < 4; ++t)
            if (t > j) v[t] = p[idx + (t - j - 1)];
        idx += 3 - j;
        float nrm = v[0]*v[0] + v[1]*v[1] + v[2]*v[2] + v[3]*v[3];
        float inv = 2.0f / nrm;
#pragma unroll
        for (int i = 0; i < 4; ++i) {
            float qv = Q[i][0]*v[0] + Q[i][1]*v[1] + Q[i][2]*v[2] + Q[i][3]*v[3];
            float s = qv * inv;
#pragma unroll
            for (int t = 0; t < 4; ++t) Q[i][t] -= s * v[t];
        }
    }
}

__launch_bounds__(256)
__global__ void q_apply_kernel(const float* __restrict__ node_rep,
                               const float* __restrict__ x, float* __restrict__ y) {
    int i = blockIdx.x * 256 + threadIdx.x;   // N*NB threads
    int n = i >> 6, b = i & 63;
    float p[6];
    const float* pp = node_rep + (size_t)n * NBP + b * 6;
#pragma unroll
    for (int t = 0; t < 6; ++t) p[t] = pp[t];
    float Q[4][4]; make_Q(p, Q);
    float4 xv = *(const float4*)(x + (size_t)n * CH + b * 4);
    float4 o;
    o.x = Q[0][0]*xv.x + Q[0][1]*xv.y + Q[0][2]*xv.z + Q[0][3]*xv.w;
    o.y = Q[1][0]*xv.x + Q[1][1]*xv.y + Q[1][2]*xv.z + Q[1][3]*xv.w;
    o.z = Q[2][0]*xv.x + Q[2][1]*xv.y + Q[2][2]*xv.z + Q[2][3]*xv.w;
    o.w = Q[3][0]*xv.x + Q[3][1]*xv.y + Q[3][2]*xv.z + Q[3][3]*xv.w;
    *(float4*)(y + (size_t)n * CH + b * 4) = o;
}

__launch_bounds__(256)
__global__ void qT_apply_kernel(const float* __restrict__ node_rep,
                                const float* __restrict__ h, float* __restrict__ out) {
    int i = blockIdx.x * 256 + threadIdx.x;
    int n = i >> 6, b = i & 63;
    float p[6];
    const float* pp = node_rep + (size_t)n * NBP + b * 6;
#pragma unroll
    for (int t = 0; t < 6; ++t) p[t] = pp[t];
    float Q[4][4]; make_Q(p, Q);
    float4 hv = *(const float4*)(h + (size_t)n * CH + b * 4);
    float4 o;  // Q^T @ hv
    o.x = Q[0][0]*hv.x + Q[1][0]*hv.y + Q[2][0]*hv.z + Q[3][0]*hv.w;
    o.y = Q[0][1]*hv.x + Q[1][1]*hv.y + Q[2][1]*hv.z + Q[3][1]*hv.w;
    o.z = Q[0][2]*hv.x + Q[1][2]*hv.y + Q[2][2]*hv.z + Q[3][2]*hv.w;
    o.w = Q[0][3]*hv.x + Q[1][3]*hv.y + Q[2][3]*hv.z + Q[3][3]*hv.w;
    *(float4*)(out + (size_t)n * CH + b * 4) = o;
}

// ---------------------------------------------------------------------------
// Spectral filter
// ---------------------------------------------------------------------------
__launch_bounds__(256)
__global__ void fact_kernel(const float* __restrict__ eigval,
                            const float* __restrict__ taus, float* __restrict__ fact) {
    int i = blockIdx.x * 256 + threadIdx.x;   // G*K*NB
    int b = i & 63, gk = i >> 6;
    fact[i] = expf(-eigval[gk] * taus[b]);
}

// hs[g,v,d] = fact[g,v,d/4] * sum_n ev[g,n,v] * y1[g,n,d]
__launch_bounds__(256)
__global__ void hs_kernel(const float* __restrict__ eigvec, const float* __restrict__ y1,
                          const float* __restrict__ fact, float* __restrict__ hs) {
    __shared__ float sev[32][33];
    __shared__ float sy[32][65];
    int g = blockIdx.x, d0 = blockIdx.y * 64;
    int tid = threadIdx.x;
    int dl = tid & 63, v0 = (tid >> 6) * 8;
    float acc[8] = {};
    const float* ev = eigvec + (size_t)g * NPG * Kk;
    const float* yy = y1 + (size_t)g * NPG * CH;
    for (int n0 = 0; n0 < NPG; n0 += 32) {
#pragma unroll
        for (int t = 0; t < 4; ++t) {
            int e = tid + t * 256, nr = e >> 5, vc = e & 31;
            sev[nr][vc] = ev[(size_t)(n0 + nr) * Kk + vc];
        }
#pragma unroll
        for (int t = 0; t < 8; ++t) {
            int e = tid + t * 256, nr = e >> 6, dc = e & 63;
            sy[nr][dc] = yy[(size_t)(n0 + nr) * CH + d0 + dc];
        }
        __syncthreads();
#pragma unroll 4
        for (int n = 0; n < 32; ++n) {
            float yv = sy[n][dl];
#pragma unroll
            for (int i = 0; i < 8; ++i) acc[i] += sev[n][v0 + i] * yv;
        }
        __syncthreads();
    }
    int bb = (d0 + dl) >> 2;
#pragma unroll
    for (int i = 0; i < 8; ++i) {
        int v = v0 + i;
        hs[((size_t)g * Kk + v) * CH + d0 + dl] =
            acc[i] * fact[((size_t)g * Kk + v) * NB + bb];
    }
}

// h2[g,n,d] = sum_v ev[g,n,v] * hs[g,v,d]   (+ bf16 copy for lin GEMM)
__launch_bounds__(256)
__global__ void reproject_kernel(const float* __restrict__ eigvec, const float* __restrict__ hs,
                                 float* __restrict__ h2, __bf16* __restrict__ h2b) {
    __shared__ float shs[32 * 256];
    __shared__ float sev[8][33];
    int g = blockIdx.x, n0 = blockIdx.y * 8;
    int tid = threadIdx.x;
    const float* hg = hs + (size_t)g * Kk * CH;
#pragma unroll
    for (int t = 0; t < 32; ++t) shs[t * 256 + tid] = hg[t * 256 + tid];
    { int nr = tid >> 5, vc = tid & 31;
      sev[nr][vc] = eigvec[((size_t)g * NPG + n0 + nr) * Kk + vc]; }
    __syncthreads();
#pragma unroll
    for (int r = 0; r < 8; ++r) {
        float acc = 0.0f;
#pragma unroll
        for (int v = 0; v < 32; ++v) acc += sev[r][v] * shs[v * 256 + tid];
        size_t off = ((size_t)g * NPG + n0 + r) * CH + tid;
        h2[off] = acc;
        h2b[off] = (__bf16)acc;
    }
}

// ---------------------------------------------------------------------------
// Host-side orchestration
// ---------------------------------------------------------------------------
static inline size_t align256(size_t x) { return (x + 255) & ~(size_t)255; }

extern "C" void kernel_launch(void* const* d_in, const int* in_sizes, int n_in,
                              void* d_out, int out_size, void* d_ws, size_t ws_size,
                              hipStream_t stream) {
    (void)in_sizes; (void)n_in; (void)out_size; (void)ws_size;
    const float* x      = (const float*)d_in[0];
    const float* eigvec = (const float*)d_in[1];
    const float* eigval = (const float*)d_in[2];
    const float* taus   = (const float*)d_in[3];
    const float* enc_w  = (const float*)d_in[4];
    const float* enc_b  = (const float*)d_in[5];
    const float* self_w = (const float*)d_in[6];
    const float* self_b = (const float*)d_in[7];
    const float* nb_w   = (const float*)d_in[8];
    const float* nb_b   = (const float*)d_in[9];
    const float* dec_w  = (const float*)d_in[10];
    const float* dec_b  = (const float*)d_in[11];
    const float* lin_w  = (const float*)d_in[12];
    const float* lin_b  = (const float*)d_in[13];
    const int*   edge   = (const int*)d_in[14];
    const int* e_src = edge;
    const int* e_dst = edge + Ee;
    float* out = (float*)d_out;

    char* ws = (char*)d_ws;
    size_t o = 0;
    auto take = [&](size_t bytes) { char* p = ws + o; o += align256(bytes); return p; };

    __bf16* xb   = (__bf16*)take((size_t)Nn * CH * 2);        // also reused as h2b
    __bf16* cat  = (__bf16*)take((size_t)Nn * 512 * 2);       // [h_bf16 | agg_bf16]; reused as h2 f32
    float*  h    = (float*) take((size_t)Nn * CH * 4);        // reused as h3
    float*  agg  = (float*) take((size_t)Nn * CH * 4);        // reused as y1
    float*  nrep = (float*) take((size_t)Nn * NBP * 4);
    __bf16* encwb = (__bf16*)take((size_t)CH * CH * 2);
    __bf16* wcat0 = (__bf16*)take((size_t)512 * CH * 2);
    __bf16* wcat1 = (__bf16*)take((size_t)512 * CH * 2);
    __bf16* decwb = (__bf16*)take((size_t)CH * NBP * 2);
    __bf16* linwb = (__bf16*)take((size_t)CH * CH * 2);
    float*  bsum0 = (float*) take(CH * 4);
    float*  bsum1 = (float*) take(CH * 4);
    float*  fact  = (float*) take((size_t)Gg * Kk * NB * 4);
    float*  hs    = (float*) take((size_t)Gg * Kk * CH * 4);

    float*  h2  = (float*)cat;   // alias (cat dead after dec GEMM)
    __bf16* h2b = xb;            // alias (xb dead after enc GEMM)
    float*  h3  = h;             // alias (h dead after dec-stage convert)
    float*  y1  = agg;           // alias (agg dead after layer-1 convert)

    auto conv = [&](const float* s, long sp, __bf16* d, long dp, long rows, long cols) {
        long tot = rows * cols;
        convert_f32_bf16_kernel<<<dim3((unsigned)((tot + 255) / 256)), dim3(256), 0, stream>>>(
            s, sp, d, dp, rows, cols);
    };
    auto gemm = [&](const __bf16* A, int lda, const __bf16* W, int ldw, const float* bias,
                    const float* res, int ldr, float* C, int ldc, __bf16* Cb, int ldcb,
                    int M, int Nd, int Kd, int act) {
        wmma_gemm_bf16_kernel<<<dim3(M / 256, Nd / 64), dim3(256), 0, stream>>>(
            A, lda, W, ldw, bias, res, ldr, C, ldc, Cb, ldcb, M, Nd, Kd, act);
    };

    // --- weight / input conversions to bf16 ---
    conv(enc_w, CH, encwb, CH, CH, CH);
    conv(self_w,            CH, wcat0,            CH, CH, CH);   // rows 0..255
    conv(nb_w,              CH, wcat0 + CH * CH,  CH, CH, CH);   // rows 256..511
    conv(self_w + CH * CH,  CH, wcat1,            CH, CH, CH);
    conv(nb_w + CH * CH,    CH, wcat1 + CH * CH,  CH, CH, CH);
    conv(dec_w, NBP, decwb, NBP, CH, NBP);
    conv(lin_w, CH, linwb, CH, CH, CH);
    conv(x, CH, xb, CH, Nn, CH);
    bias_sum_kernel<<<1, 256, 0, stream>>>(self_b, nb_b, bsum0, CH);
    bias_sum_kernel<<<1, 256, 0, stream>>>(self_b + CH, nb_b + CH, bsum1, CH);

    // --- encoder: h = gelu(x @ enc_w + enc_b); bf16(h) -> cat[:,0:256] ---
    gemm(xb, CH, encwb, CH, enc_b, nullptr, 0, h, CH, cat, 512, Nn, CH, CH, 1);

    const long nch = (long)Nn * CH;
    const unsigned zb = (unsigned)((nch + 255) / 256);
    const unsigned sb = (unsigned)(((long)Ee * 64) / 256);

    // --- GNN layer 0 ---
    zero_f32_kernel<<<zb, 256, 0, stream>>>(agg, nch);
    scatter_add_kernel<<<sb, 256, 0, stream>>>(e_src, e_dst, h, agg);
    conv(agg, CH, cat + CH, 512, Nn, CH);
    gemm(cat, 512, wcat0, CH, bsum0, h, CH, h, CH, nullptr, 0, Nn, CH, 512, 1);

    // --- GNN layer 1 ---
    conv(h, CH, cat, 512, Nn, CH);
    zero_f32_kernel<<<zb, 256, 0, stream>>>(agg, nch);
    scatter_add_kernel<<<sb, 256, 0, stream>>>(e_src, e_dst, h, agg);
    conv(agg, CH, cat + CH, 512, Nn, CH);
    gemm(cat, 512, wcat1, CH, bsum1, h, CH, h, CH, nullptr, 0, Nn, CH, 512, 1);

    // --- decoder: node_rep = h @ dec_w + dec_b ---
    conv(h, CH, cat, 512, Nn, CH);
    gemm(cat, 512, decwb, NBP, dec_b, nullptr, 0, nrep, NBP, nullptr, 0, Nn, NBP, CH, 0);

    // --- bundle transform: y1 = blockdiag(Q) @ x ---
    q_apply_kernel<<<dim3((Nn * NB) / 256), dim3(256), 0, stream>>>(nrep, x, y1);

    // --- spectral filter ---
    fact_kernel<<<dim3((Gg * Kk * NB) / 256), dim3(256), 0, stream>>>(eigval, taus, fact);
    hs_kernel<<<dim3(Gg, CH / 64), dim3(256), 0, stream>>>(eigvec, y1, fact, hs);
    reproject_kernel<<<dim3(Gg, NPG / 8), dim3(256), 0, stream>>>(eigvec, hs, h2, h2b);

    // --- lin: h3 = h2 @ lin_w + lin_b ---
    gemm(h2b, CH, linwb, CH, lin_b, nullptr, 0, h3, CH, nullptr, 0, Nn, CH, CH, 0);

    // --- output: out = blockdiag(Q)^T @ h3 ---
    qT_apply_kernel<<<dim3((Nn * NB) / 256), dim3(256), 0, stream>>>(nrep, h3, out);
}
